// GeometricCrossAttention_11261404250495
// MI455X (gfx1250) — compile-verified, matrix operand order checked by reading, not run on hardware
//
#include <hip/hip_runtime.h>
#include <hip/hip_bf16.h>
#include <math.h>

typedef __attribute__((ext_vector_type(16))) _Float16 v16h;
typedef __attribute__((ext_vector_type(2)))  _Float16 h2;
typedef __attribute__((ext_vector_type(8)))  float    v8f;

#define L_DIM 1024
#define P_DIM 4096
#define D_DIM 512
#define H_NUM 8
#define DH    64
#define R_DIM 50

__device__ inline v8f wmma_f16(v16h a, v16h b, v8f c) {
  return __builtin_amdgcn_wmma_f32_16x16x32_f16(
      /*neg_a=*/false, a, /*neg_b=*/false, b,
      /*c_mod=*/(short)0, c, /*reuse_a=*/false, /*reuse_b=*/false);
}

// Load one 16x32 fragment (A or B) from a row-major f16 row with contiguous K.
// ISA 7.12.2 lane layout: element e: v=e>>1,h=e&1 ->
//   k = ((v>>2)<<4) + kgrp + ((v&3)<<1) + h, kgrp = (lane&16)?8:0.
__device__ inline v16h load_frag_f16(const _Float16* __restrict__ row, int kgrp) {
  v16h f;
#pragma unroll
  for (int v = 0; v < 8; ++v) {
    const int k = ((v >> 2) << 4) + kgrp + ((v & 3) << 1);
    const h2 t = *(const h2*)(row + k);
    f[2 * v]     = t.x;
    f[2 * v + 1] = t.y;
  }
  return f;
}

// ---------------------------------------------------------------------------
// Projection GEMM: C[M,N] = (A[M,K]*W[N,K]^T + bias[N]) * oscale
// f32 operands -> f16 on load, f32 WMMA accumulate.
// Output: f16 (optionally transposed, for Vt) or f32.
// One 16x16 tile per wave, 4 waves per block (16x64 strip).
// ---------------------------------------------------------------------------
__global__ __launch_bounds__(128)
void proj_gemm(const float* __restrict__ A, const float* __restrict__ W,
               const float* __restrict__ bias, float* __restrict__ Cf,
               _Float16* __restrict__ Ch, int M, int N, int Kd,
               float oscale, int transposeOut) {
  const int wave = threadIdx.x >> 5;
  const int lane = threadIdx.x & 31;
  const int tileM = blockIdx.x * 16;
  const int tileN = blockIdx.y * 64 + wave * 16;
  const int mrow = tileM + (lane & 15);
  const int ncol = tileN + (lane & 15);
  const int kgrp = (lane & 16) ? 8 : 0;

  const float* __restrict__ Arow = A + (size_t)mrow * Kd;
  const float* __restrict__ Wrow = W + (size_t)ncol * Kd;

  v8f acc = {};
#pragma unroll 4
  for (int k0 = 0; k0 < Kd; k0 += 32) {
    v16h a, b;
#pragma unroll
    for (int v = 0; v < 8; ++v) {
      const int k = k0 + ((v >> 2) << 4) + kgrp + ((v & 3) << 1);
      const float2 av = *(const float2*)(Arow + k);
      const float2 bv = *(const float2*)(Wrow + k);
      a[2 * v]     = (_Float16)av.x;
      a[2 * v + 1] = (_Float16)av.y;
      b[2 * v]     = (_Float16)bv.x;
      b[2 * v + 1] = (_Float16)bv.y;
    }
    acc = wmma_f16(a, b, acc);
  }

  const float bn = bias ? bias[ncol] : 0.0f;
  const int mbase = tileM + kgrp;
#pragma unroll
  for (int r = 0; r < 8; ++r) {
    const float val = (acc[r] + bn) * oscale;
    if (Ch) {
      if (transposeOut) Ch[(size_t)ncol * M + (mbase + r)] = (_Float16)val;
      else              Ch[(size_t)(mbase + r) * N + ncol] = (_Float16)val;
    } else {
      Cf[(size_t)(mbase + r) * N + ncol] = val;
    }
  }
}

// ---------------------------------------------------------------------------
// Scores: S[l,h,p] (f16) = Qh[l,h,:] . Kh[p,h,:]  (Qh pre-scaled by 1/sqrt(d))
//                        + rbf[l,p,:] . Wr[h,:] + br[h]
// Block = 16 l x 128 p tile, 256 threads (8 waves).
// Phase 1: bias via WMMA over (l,p)-pairs: M=2048, K=50 (padded 64), N=8
//          (padded 16); cross_rbf streamed exactly once, with prefetch.
// Phase 2: per-head Q.K^T WMMA with C preloaded from the LDS bias tile.
// ---------------------------------------------------------------------------
__global__ __launch_bounds__(256)
void scores_kernel(const _Float16* __restrict__ Qh, const _Float16* __restrict__ Kh,
                   const float* __restrict__ rbf, const float* __restrict__ Wr,
                   const float* __restrict__ br, _Float16* __restrict__ S) {
  extern __shared__ float bias_s[];            // [8 heads][2048 pairs] = 64 KB
  const int tid  = threadIdx.x;
  const int wave = tid >> 5;
  const int lane = tid & 31;
  const int tileL = blockIdx.x * 16;
  const int tileP = blockIdx.y * 128;
  const int hi8  = (lane & 16) ? 8 : 0;
  const int kgrp = hi8;
  const int l15  = lane & 15;

  // ---- Phase 1: bias tile via WMMA -------------------------------------
  // B fragments built from Wr (zero-padded: k>=50 or n>=8 -> 0).
  v16h bw0, bw1;
#pragma unroll
  for (int v = 0; v < 8; ++v) {
    const int kb = ((v >> 2) << 4) + kgrp + ((v & 3) << 1);   // 0..31
    const int kc = kb + 32;                                    // 32..63
    const float x0 = (l15 < 8) ? Wr[l15 * R_DIM + kb] : 0.0f;  // kb<50 always
    const float x1 = (l15 < 8) ? Wr[l15 * R_DIM + kb + 1] : 0.0f;
    const float y0 = (l15 < 8 && kc < R_DIM)     ? Wr[l15 * R_DIM + kc]     : 0.0f;
    const float y1 = (l15 < 8 && kc + 1 < R_DIM) ? Wr[l15 * R_DIM + kc + 1] : 0.0f;
    bw0[2 * v] = (_Float16)x0; bw0[2 * v + 1] = (_Float16)x1;
    bw1[2 * v] = (_Float16)y0; bw1[2 * v + 1] = (_Float16)y1;
  }
  const float brv = (l15 < 8) ? br[l15] : 0.0f;

  for (int t = 0; t < 16; ++t) {
    const int mtile = wave * 16 + t;           // 0..127 (16 pairs each)
    const int q = mtile * 16 + l15;            // pair index 0..2047
    const int ll = q >> 7, pp = q & 127;
    const float* __restrict__ rrow =
        rbf + ((size_t)(tileL + ll) * P_DIM + tileP + pp) * R_DIM;
    if (t + 2 < 16) __builtin_prefetch(rrow + 32 * R_DIM, 0, 0);

    v16h a0, a1;
#pragma unroll
    for (int v = 0; v < 8; ++v) {
      const int kb = ((v >> 2) << 4) + kgrp + ((v & 3) << 1);  // <= 31 < 50
      const int kc = kb + 32;
      const float2 t0 = *(const float2*)(rrow + kb);
      float2 t1 = make_float2(0.0f, 0.0f);
      if (kc < R_DIM) t1 = *(const float2*)(rrow + kc);        // pair (48,49) ok
      a0[2 * v] = (_Float16)t0.x; a0[2 * v + 1] = (_Float16)t0.y;
      a1[2 * v] = (_Float16)t1.x; a1[2 * v + 1] = (_Float16)t1.y;
    }
    v8f acc = {};
    acc = wmma_f16(a0, bw0, acc);
    acc = wmma_f16(a1, bw1, acc);
    if (l15 < 8) {                              // N-cols 8..15 are padding
#pragma unroll
      for (int r = 0; r < 8; ++r)
        bias_s[l15 * 2048 + mtile * 16 + r + hi8] = acc[r] + brv;
    }
  }
  __syncthreads();

  // ---- Phase 2: per-head scores WMMA, C preloaded with bias ------------
  const int mrow   = tileL + l15;
  const int plocal = wave * 16 + l15;
  const int pcol   = tileP + plocal;
  for (int h = 0; h < 8; ++h) {
    const _Float16* __restrict__ Ar = Qh + (size_t)mrow * D_DIM + h * DH;
    const _Float16* __restrict__ Br = Kh + (size_t)pcol * D_DIM + h * DH;
    v8f acc;
#pragma unroll
    for (int r = 0; r < 8; ++r)
      acc[r] = bias_s[h * 2048 + (r + hi8) * 128 + plocal];
    v16h a = load_frag_f16(Ar, kgrp);
    v16h b = load_frag_f16(Br, kgrp);
    acc = wmma_f16(a, b, acc);
    a = load_frag_f16(Ar + 32, kgrp);
    b = load_frag_f16(Br + 32, kgrp);
    acc = wmma_f16(a, b, acc);
#pragma unroll
    for (int r = 0; r < 8; ++r)
      S[((size_t)(tileL + r + hi8) * H_NUM + h) * P_DIM + pcol] = (_Float16)acc[r];
  }
}

// ---------------------------------------------------------------------------
// Softmax over P per (l,h), in-place on f16 S; attn-mean from f32 values.
// ---------------------------------------------------------------------------
__global__ __launch_bounds__(256)
void softmax_kernel(_Float16* __restrict__ S, float* __restrict__ attn_mean) {
  __shared__ float red[256];
  const int l = blockIdx.x, tid = threadIdx.x;
  float meanacc[16];
#pragma unroll
  for (int i = 0; i < 16; ++i) meanacc[i] = 0.0f;

  for (int h = 0; h < H_NUM; ++h) {
    _Float16* __restrict__ row = S + ((size_t)l * H_NUM + h) * P_DIM;
    float v[16];
    float m = -3.0e38f;
#pragma unroll
    for (int i = 0; i < 16; ++i) {
      v[i] = (float)row[i * 256 + tid];
      m = fmaxf(m, v[i]);
    }
    red[tid] = m;
    __syncthreads();
    for (int st = 128; st > 0; st >>= 1) {
      if (tid < st) red[tid] = fmaxf(red[tid], red[tid + st]);
      __syncthreads();
    }
    m = red[0];
    __syncthreads();
    float sum = 0.0f;
#pragma unroll
    for (int i = 0; i < 16; ++i) {
      v[i] = __expf(v[i] - m);
      sum += v[i];
    }
    red[tid] = sum;
    __syncthreads();
    for (int st = 128; st > 0; st >>= 1) {
      if (tid < st) red[tid] += red[tid + st];
      __syncthreads();
    }
    const float inv = 1.0f / red[0];
    __syncthreads();
#pragma unroll
    for (int i = 0; i < 16; ++i) {
      const float a = v[i] * inv;
      meanacc[i] += a;
      row[i * 256 + tid] = (_Float16)a;
    }
  }
#pragma unroll
  for (int i = 0; i < 16; ++i)
    attn_mean[(size_t)l * P_DIM + i * 256 + tid] = meanacc[i] * 0.125f;
}

// ---------------------------------------------------------------------------
// attn @ V per head: out[l, h*64+n] = sum_p attn[l,h,p] * Vt[h*64+n, p]
// Pure WMMA, K = 4096. Block: 4 waves over the 64-wide head slice.
// ---------------------------------------------------------------------------
__global__ __launch_bounds__(128)
void pv_gemm(const _Float16* __restrict__ S, const _Float16* __restrict__ Vt,
             float* __restrict__ out) {
  const int wave = threadIdx.x >> 5;
  const int lane = threadIdx.x & 31;
  const int tileL = blockIdx.x * 16;
  const int h = blockIdx.y;
  const int kgrp = (lane & 16) ? 8 : 0;
  const int n = wave * 16 + (lane & 15);      // 0..63 within head
  const int d = h * DH + n;

  const _Float16* __restrict__ Ar =
      S + ((size_t)(tileL + (lane & 15)) * H_NUM + h) * P_DIM;
  const _Float16* __restrict__ Br = Vt + (size_t)d * P_DIM;

  v8f acc = {};
#pragma unroll 4
  for (int k0 = 0; k0 < P_DIM; k0 += 32) {
    const v16h a = load_frag_f16(Ar + k0, kgrp);
    const v16h b = load_frag_f16(Br + k0, kgrp);
    acc = wmma_f16(a, b, acc);
  }
  const int mbase = tileL + kgrp;
#pragma unroll
  for (int r = 0; r < 8; ++r)
    out[(size_t)(mbase + r) * D_DIM + d] = acc[r];
}

// ---------------------------------------------------------------------------
// Residual + LayerNorm: y = LN(query + O2) * gamma + beta
// ---------------------------------------------------------------------------
__global__ __launch_bounds__(256)
void ln_kernel(const float* __restrict__ query, const float* __restrict__ O2,
               const float* __restrict__ gamma, const float* __restrict__ beta,
               float* __restrict__ y) {
  __shared__ float redsum[256];
  __shared__ float redsq[256];
  const int l = blockIdx.x, tid = threadIdx.x;
  const size_t base = (size_t)l * D_DIM;
  const float x0 = query[base + tid]       + O2[base + tid];
  const float x1 = query[base + 256 + tid] + O2[base + 256 + tid];
  redsum[tid] = x0 + x1;
  redsq[tid]  = x0 * x0 + x1 * x1;
  __syncthreads();
  for (int st = 128; st > 0; st >>= 1) {
    if (tid < st) {
      redsum[tid] += redsum[tid + st];
      redsq[tid]  += redsq[tid + st];
    }
    __syncthreads();
  }
  const float mu   = redsum[0] * (1.0f / 512.0f);
  const float var  = redsq[0] * (1.0f / 512.0f) - mu * mu;
  const float rstd = rsqrtf(var + 1e-5f);
  y[base + tid]       = (x0 - mu) * rstd * gamma[tid]       + beta[tid];
  y[base + 256 + tid] = (x1 - mu) * rstd * gamma[tid + 256] + beta[tid + 256];
}

// ---------------------------------------------------------------------------
extern "C" void kernel_launch(void* const* d_in, const int* in_sizes, int n_in,
                              void* d_out, int out_size, void* d_ws, size_t ws_size,
                              hipStream_t stream) {
  (void)in_sizes; (void)n_in; (void)out_size; (void)ws_size;

  const float* query = (const float*)d_in[0];   // [1024,512]
  const float* key   = (const float*)d_in[1];   // [4096,512]
  const float* value = (const float*)d_in[2];   // [4096,512]
  const float* rbf   = (const float*)d_in[3];   // [1024,4096,50]
  const float* Wq    = (const float*)d_in[4];
  const float* bq    = (const float*)d_in[5];
  const float* Wk    = (const float*)d_in[6];
  const float* bk    = (const float*)d_in[7];
  const float* Wv    = (const float*)d_in[8];
  const float* bv    = (const float*)d_in[9];
  const float* Wr    = (const float*)d_in[10];  // [8,50]
  const float* br    = (const float*)d_in[11];  // [8]
  const float* Wo    = (const float*)d_in[12];
  const float* bo    = (const float*)d_in[13];
  const float* gamma = (const float*)d_in[14];
  const float* beta  = (const float*)d_in[15];

  float* y_out  = (float*)d_out;                          // [1024,512]
  float* am_out = (float*)d_out + (size_t)L_DIM * D_DIM;  // [1024,4096]

  char* ws = (char*)d_ws;
  _Float16* Qh = (_Float16*)ws;                                   //  1 MB
  _Float16* Kh = (_Float16*)(ws + (1u << 20));                    //  4 MB
  _Float16* Vt = (_Float16*)(ws + (5u << 20));                    //  4 MB
  _Float16* Sb = (_Float16*)(ws + (9u << 20));                    // 64 MB
  float*    Ow  = (float*)(ws + (73u << 20));                     //  2 MB
  float*    O2w = (float*)(ws + (75u << 20));                     //  2 MB

  // QKV projections (WMMA, f16 outputs; Q pre-scaled by 1/sqrt(64)=0.125)
  proj_gemm<<<dim3(L_DIM / 16, D_DIM / 64), 128, 0, stream>>>(
      query, Wq, bq, nullptr, Qh, L_DIM, D_DIM, D_DIM, 0.125f, 0);
  proj_gemm<<<dim3(P_DIM / 16, D_DIM / 64), 128, 0, stream>>>(
      key, Wk, bk, nullptr, Kh, P_DIM, D_DIM, D_DIM, 1.0f, 0);
  proj_gemm<<<dim3(P_DIM / 16, D_DIM / 64), 128, 0, stream>>>(
      value, Wv, bv, nullptr, Vt, P_DIM, D_DIM, D_DIM, 1.0f, 1);

  // Scores: WMMA rbf-bias (stream cross_rbf once) + WMMA Q.K^T
  const size_t smem = (size_t)H_NUM * 2048 * sizeof(float);       // 64 KB
  (void)hipFuncSetAttribute((const void*)scores_kernel,
                            hipFuncAttributeMaxDynamicSharedMemorySize,
                            (int)smem);
  scores_kernel<<<dim3(L_DIM / 16, P_DIM / 128), 256, smem, stream>>>(
      Qh, Kh, rbf, Wr, br, Sb);

  // Softmax over P + attn mean
  softmax_kernel<<<L_DIM, 256, 0, stream>>>(Sb, am_out);

  // attn @ V (WMMA, K=4096)
  pv_gemm<<<dim3(L_DIM / 16, H_NUM), 128, 0, stream>>>(Sb, Vt, Ow);

  // Output projection (WMMA) + residual LayerNorm
  proj_gemm<<<dim3(L_DIM / 16, D_DIM / 64), 128, 0, stream>>>(
      Ow, Wo, bo, O2w, nullptr, L_DIM, D_DIM, D_DIM, 1.0f, 0);
  ln_kernel<<<L_DIM, 256, 0, stream>>>(query, O2w, gamma, beta, y_out);
}
